// RetNet_31653908971898
// MI455X (gfx1250) — matrix operand order repaired and checked by reading, not compiled, not run on hardware
//
#include <hip/hip_runtime.h>

// ---------------------------------------------------------------------------
// RetNet block for MI455X (gfx1250, wave32, WMMA).
// All matmuls run as bf16 WMMA (v_wmma_f32_16x16x32_bf16) with fp32 accum:
// fp32 WMMA is 16x16x4 (8x lower ceiling), and the problem is compute-bound
// (~146 GFLOP vs ~100MB of traffic -> HBM floor ~4.5us at 23.3 TB/s).
// ---------------------------------------------------------------------------

#define D_MODEL 1024
#define SEQ     2048
#define BATCH   2
#define ROWS    (BATCH * SEQ)   // 4096
#define NHEAD   16
#define DH      64
#define HID     2730
#define HID2    2816            // HID padded to a multiple of 128 (zero-filled)

typedef __attribute__((ext_vector_type(16))) __bf16 v16bf;
typedef __attribute__((ext_vector_type(8)))  __bf16 v8bf;
typedef __attribute__((ext_vector_type(8)))  float  v8f;

static __device__ __forceinline__ __bf16 f2bf(float f) {
  union { float f; unsigned u; } x; x.f = f;
  unsigned r = (x.u + 0x7fffu + ((x.u >> 16) & 1u)) >> 16;  // round-nearest-even
  unsigned short s = (unsigned short)r;
  return __builtin_bit_cast(__bf16, s);
}

static __device__ __forceinline__ v8f wmma_bf16(v16bf a, v16bf b, v8f c) {
  return __builtin_amdgcn_wmma_f32_16x16x32_bf16(false, a, false, b, (short)0, c,
                                                 false, false);
}

// A-fragment (16x32 bf16): per lane two 16B chunks, K {0..15} (lanes<16)
// or K {16..31} (lanes>=16). Assembled from two v8bf loads.
static __device__ __forceinline__ v16bf frag2(const __bf16* p0, const __bf16* p1) {
  v8bf lo = *(const v8bf*)p0;
  v8bf hi = *(const v8bf*)p1;
  v16bf r;
#pragma unroll
  for (int i = 0; i < 8; ++i) { r[i] = lo[i]; r[8 + i] = hi[i]; }
  return r;
}

// B-fragment (32x16 bf16): per lane one contiguous 32B chunk of K values from
// the transposed (column-major-of-W) weight row. Requires 32B alignment.
static __device__ __forceinline__ v16bf frag1(const __bf16* p) {
  return *(const v16bf*)p;
}

// ---------------------------------------------------------------------------
// fp32 -> bf16 transposed weight conversion: dst[n][k] = src[k*N + n] * scale,
// zero-filled outside (K,N) so padded GEMM dims are inert.
// ---------------------------------------------------------------------------
__global__ __launch_bounds__(256)
void cvtT_kernel(const float* __restrict__ src, __bf16* __restrict__ dst,
                 int K, int N, int Kout, float scale, long total) {
  long idx = (long)blockIdx.x * 256 + threadIdx.x;
  if (idx >= total) return;
  int n = (int)(idx / Kout);
  int k = (int)(idx % Kout);
  float v = 0.f;
  if (k < K && n < N) v = src[(size_t)k * N + n] * scale;
  dst[idx] = f2bf(v);
}

// ---------------------------------------------------------------------------
// LayerNorm(row of 1024) -> bf16. One block per row.
// ---------------------------------------------------------------------------
__global__ __launch_bounds__(256)
void ln_kernel(const float* __restrict__ x, const float* __restrict__ g,
               const float* __restrict__ b, __bf16* __restrict__ out) {
  int row = blockIdx.x, tid = threadIdx.x;
  const float* xr = x + (size_t)row * D_MODEL;
  float v[4]; float s1 = 0.f, s2 = 0.f;
#pragma unroll
  for (int i = 0; i < 4; ++i) {
    float t = xr[tid + i * 256]; v[i] = t; s1 += t; s2 += t * t;
  }
  __shared__ float red[256];
  red[tid] = s1; __syncthreads();
  for (int off = 128; off; off >>= 1) { if (tid < off) red[tid] += red[tid + off]; __syncthreads(); }
  float tot1 = red[0]; __syncthreads();
  red[tid] = s2; __syncthreads();
  for (int off = 128; off; off >>= 1) { if (tid < off) red[tid] += red[tid + off]; __syncthreads(); }
  float tot2 = red[0];
  float mean = tot1 * (1.f / D_MODEL);
  float var  = tot2 * (1.f / D_MODEL) - mean * mean;
  float inv  = rsqrtf(var + 1e-5f);
  __bf16* orow = out + (size_t)row * D_MODEL;
#pragma unroll
  for (int i = 0; i < 4; ++i) {
    int c = tid + i * 256;
    orow[c] = f2bf((v[i] - mean) * inv * g[c] + b[c]);
  }
}

// ---------------------------------------------------------------------------
// WMMA GEMM: C[M,N] = A[M,K](bf16) x W(bf16, stored transposed WT[N][K]).
// Block tile 128x128, 8 waves (4Mx2N), wave tile 32x64 = 2x4 WMMA C tiles.
// Epilogues: bf16 store / f32 store / f32 residual-add store.
// ---------------------------------------------------------------------------
enum { EPI_BF16 = 0, EPI_F32 = 1, EPI_RES = 2 };

template <int EPI>
__global__ __launch_bounds__(256)
void gemm_kernel(const __bf16* __restrict__ A, const __bf16* __restrict__ WT,
                 const float* __restrict__ res, void* __restrict__ outp,
                 int N, int K) {
  int tid = threadIdx.x, lane = tid & 31, w = tid >> 5;
  int wm = w & 3, wn = w >> 2;
  int rowBase = blockIdx.y * 128 + wm * 32;
  int colBase = blockIdx.x * 128 + wn * 64;
  int l16  = lane & 15;
  int aoff0 = (lane < 16) ? 0 : 16;
  int aoff1 = (lane < 16) ? 8 : 24;
  int boff  = (lane < 16) ? 0 : 16;

  const __bf16* arow[2];
  const __bf16* brow[4];
#pragma unroll
  for (int i = 0; i < 2; ++i) arow[i] = A + (size_t)(rowBase + i * 16 + l16) * K;
#pragma unroll
  for (int j = 0; j < 4; ++j) brow[j] = WT + (size_t)(colBase + j * 16 + l16) * K + boff;

  v8f acc[2][4] = {};

  for (int k0 = 0; k0 < K; k0 += 32) {
    v16bf a[2], b[4];
#pragma unroll
    for (int i = 0; i < 2; ++i)
      a[i] = frag2(arow[i] + k0 + aoff0, arow[i] + k0 + aoff1);
#pragma unroll
    for (int j = 0; j < 4; ++j) b[j] = frag1(brow[j] + k0);
#pragma unroll
    for (int i = 0; i < 2; ++i)
#pragma unroll
      for (int j = 0; j < 4; ++j) acc[i][j] = wmma_bf16(a[i], b[j], acc[i][j]);
  }

  int rHalf = (lane < 16) ? 0 : 8;
#pragma unroll
  for (int i = 0; i < 2; ++i)
#pragma unroll
    for (int j = 0; j < 4; ++j)
#pragma unroll
      for (int r = 0; r < 8; ++r) {
        int row = rowBase + i * 16 + r + rHalf;
        int col = colBase + j * 16 + l16;
        size_t o = (size_t)row * N + col;
        float v = acc[i][j][r];
        if (EPI == EPI_BF16)     ((__bf16*)outp)[o] = f2bf(v);
        else if (EPI == EPI_F32) ((float*)outp)[o] = v;
        else                     ((float*)outp)[o] = res[o] + v;
      }
}

// ---------------------------------------------------------------------------
// SwiGLU dual-GEMM: hmid = bf16( silu(A@W1) * (A@W2) ), block 128x64, 8 waves.
// ---------------------------------------------------------------------------
__global__ __launch_bounds__(256)
void swiglu_kernel(const __bf16* __restrict__ A, const __bf16* __restrict__ W1T,
                   const __bf16* __restrict__ W2T, __bf16* __restrict__ out,
                   int N, int K) {
  int tid = threadIdx.x, lane = tid & 31, w = tid >> 5;
  int wm = w & 3, wn = w >> 2;                   // 4 x 2 waves
  int rowBase = blockIdx.y * 128 + wm * 32;
  int colBase = blockIdx.x * 64 + wn * 32;
  int l16  = lane & 15;
  int aoff0 = (lane < 16) ? 0 : 16;
  int aoff1 = (lane < 16) ? 8 : 24;
  int boff  = (lane < 16) ? 0 : 16;

  const __bf16* arow[2];
  const __bf16 *b1row[2], *b2row[2];
#pragma unroll
  for (int i = 0; i < 2; ++i) arow[i] = A + (size_t)(rowBase + i * 16 + l16) * K;
#pragma unroll
  for (int j = 0; j < 2; ++j) {
    size_t o = (size_t)(colBase + j * 16 + l16) * K + boff;
    b1row[j] = W1T + o; b2row[j] = W2T + o;
  }

  v8f a1[2][2] = {}, a2[2][2] = {};

  for (int k0 = 0; k0 < K; k0 += 32) {
    v16bf a[2];
#pragma unroll
    for (int i = 0; i < 2; ++i)
      a[i] = frag2(arow[i] + k0 + aoff0, arow[i] + k0 + aoff1);
#pragma unroll
    for (int j = 0; j < 2; ++j) {
      v16bf b1 = frag1(b1row[j] + k0);
      v16bf b2 = frag1(b2row[j] + k0);
#pragma unroll
      for (int i = 0; i < 2; ++i) {
        a1[i][j] = wmma_bf16(a[i], b1, a1[i][j]);
        a2[i][j] = wmma_bf16(a[i], b2, a2[i][j]);
      }
    }
  }

  int rHalf = (lane < 16) ? 0 : 8;
#pragma unroll
  for (int i = 0; i < 2; ++i)
#pragma unroll
    for (int j = 0; j < 2; ++j)
#pragma unroll
      for (int r = 0; r < 8; ++r) {
        int row = rowBase + i * 16 + r + rHalf;
        int col = colBase + j * 16 + l16;
        float x1 = a1[i][j][r], x2 = a2[i][j][r];
        float sg = x1 / (1.f + __expf(-x1));
        out[(size_t)row * N + col] = f2bf(sg * x2);
      }
}

// ---------------------------------------------------------------------------
// Retention core. Grid (S/128, H, B); 8 waves own 16-row strips of a 128-row
// query tile. Per 32-key step: scores = QK^T (2 WMMA over dh=64), decay+mask
// in registers, bf16 scores -> per-wave LDS (A-source layout), V staged
// transposed in LDS (stride 40: 16B-aligned, conflict-free), 4 WMMAs into o.
// Epilogue: per-row GroupNorm(dh) via 16-lane shfl_xor, SiLU gate, bf16 out.
// ---------------------------------------------------------------------------
__global__ __launch_bounds__(256)
void retention_kernel(const __bf16* __restrict__ qb, const __bf16* __restrict__ kb,
                      const __bf16* __restrict__ vb, const float* __restrict__ xg,
                      __bf16* __restrict__ og) {
  __shared__ __align__(16) __bf16 vT[64 * 40];       // [dh][key], stride 40
  __shared__ __align__(16) __bf16 sc[8][16 * 32];    // per-wave decayed scores

  int tid = threadIdx.x, lane = tid & 31, w = tid >> 5;
  int l16 = lane & 15;
  int n0 = blockIdx.x * 128;
  int h  = blockIdx.y;
  int bI = blockIdx.z;
  size_t rowBase = (size_t)bI * SEQ;
  int hoff = h * DH;
  float log2g = __log2f(1.0f - exp2f(-5.0f - (float)h));

  int aoff0 = (lane < 16) ? 0 : 16;
  int aoff1 = (lane < 16) ? 8 : 24;
  int boff  = (lane < 16) ? 0 : 16;
  int rHalf = (lane < 16) ? 0 : 8;

  int rowQ = n0 + w * 16 + l16;
  const __bf16* qrow = qb + (rowBase + rowQ) * D_MODEL + hoff;   // q has 0.125 folded in

  v8f acc[4] = {};

  int mEnd = n0 + 128;
  for (int m0 = 0; m0 < mEnd; m0 += 32) {
    // scores for two 16-key sub-tiles (global loads only, no LDS dep yet)
    v8f s[2] = {};
#pragma unroll
    for (int sub = 0; sub < 2; ++sub) {
      int mB = m0 + sub * 16;
      const __bf16* krow = kb + (rowBase + mB + l16) * D_MODEL + hoff + boff;
#pragma unroll
      for (int dstep = 0; dstep < DH; dstep += 32) {
        v16bf a = frag2(qrow + dstep + aoff0, qrow + dstep + aoff1);
        v16bf b = frag1(krow + dstep);
        s[sub] = wmma_bf16(a, b, s[sub]);
      }
    }
    // decay + causal mask -> bf16 score tile in this wave's LDS region
#pragma unroll
    for (int sub = 0; sub < 2; ++sub)
#pragma unroll
      for (int r = 0; r < 8; ++r) {
        int nAbs = n0 + w * 16 + r + rHalf;
        int mAbs = m0 + sub * 16 + l16;
        int d = nAbs - mAbs;
        float v = (d < 0) ? 0.f : s[sub][r] * exp2f((float)d * log2g);
        sc[w][(r + rHalf) * 32 + sub * 16 + l16] = f2bf(v);
      }
    // stage 32x64 V tile transposed: vT[d][key]
    {
      int r = tid & 31, dg = tid >> 5;
      const __bf16* vrow = vb + (rowBase + m0 + r) * D_MODEL + hoff + dg * 8;
      v8bf vv = *(const v8bf*)vrow;
#pragma unroll
      for (int i = 0; i < 8; ++i) vT[(dg * 8 + i) * 40 + r] = vv[i];
    }
    __syncthreads();
    // o += decayed_scores(16x32) @ V(32x64)
    v16bf a2 = frag2(&sc[w][l16 * 32 + aoff0], &sc[w][l16 * 32 + aoff1]);
#pragma unroll
    for (int t = 0; t < 4; ++t) {
      const __bf16* p = &vT[(t * 16 + l16) * 40 + boff];
      v16bf b = frag2(p, p + 8);       // 16B-aligned halves (row stride 80B)
      acc[t] = wmma_bf16(a2, b, acc[t]);
    }
    __syncthreads();
  }

  // GroupNorm over dh=64 per row (row lives across 16 lanes of a half-wave),
  // then SiLU(xg) gate, store bf16 for the wo projection.
#pragma unroll
  for (int r = 0; r < 8; ++r) {
    float s1 = 0.f, s2 = 0.f;
#pragma unroll
    for (int t = 0; t < 4; ++t) { float v = acc[t][r]; s1 += v; s2 += v * v; }
#pragma unroll
    for (int m = 1; m < 16; m <<= 1) {
      s1 += __shfl_xor(s1, m, 32);
      s2 += __shfl_xor(s2, m, 32);
    }
    float mean = s1 * (1.f / DH);
    float var  = s2 * (1.f / DH) - mean * mean;
    float inv  = rsqrtf(var + 1e-6f);
    int rowA = n0 + w * 16 + r + rHalf;
    size_t base = (rowBase + rowA) * D_MODEL + hoff;
#pragma unroll
    for (int t = 0; t < 4; ++t) {
      int c = t * 16 + l16;
      float gv = xg[base + c];
      float gate = gv / (1.f + __expf(-gv));
      og[base + c] = f2bf((acc[t][r] - mean) * inv * gate);
    }
  }
}

// ---------------------------------------------------------------------------
extern "C" void kernel_launch(void* const* d_in, const int* in_sizes, int n_in,
                              void* d_out, int out_size, void* d_ws, size_t ws_size,
                              hipStream_t stream) {
  const float* x    = (const float*)d_in[0];
  const float* ln1g = (const float*)d_in[1];
  const float* ln1b = (const float*)d_in[2];
  const float* ln2g = (const float*)d_in[3];
  const float* ln2b = (const float*)d_in[4];
  const float* wq   = (const float*)d_in[5];
  const float* wk   = (const float*)d_in[6];
  const float* wv   = (const float*)d_in[7];
  const float* wg   = (const float*)d_in[8];
  const float* wo   = (const float*)d_in[9];
  const float* w1   = (const float*)d_in[10];
  const float* w2   = (const float*)d_in[11];
  const float* w3   = (const float*)d_in[12];
  float* out = (float*)d_out;

  char* ws = (char*)d_ws;
  size_t off = 0;
  auto take = [&](size_t bytes) -> char* {
    char* p = ws + off;
    off += (bytes + 255) & ~(size_t)255;
    return p;
  };

  __bf16* wqT = (__bf16*)take((size_t)D_MODEL * D_MODEL * 2);
  __bf16* wkT = (__bf16*)take((size_t)D_MODEL * D_MODEL * 2);
  __bf16* wvT = (__bf16*)take((size_t)D_MODEL * D_MODEL * 2);
  __bf16* wgT = (__bf16*)take((size_t)D_MODEL * D_MODEL * 2);
  __bf16* woT = (__bf16*)take((size_t)D_MODEL * D_MODEL * 2);
  __bf16* w1T = (__bf16*)take((size_t)HID2 * D_MODEL * 2);
  __bf16* w2T = (__bf16*)take((size_t)HID2 * D_MODEL * 2);
  __bf16* w3T = (__bf16*)take((size_t)D_MODEL * HID2 * 2);
  __bf16* xn  = (__bf16*)take((size_t)ROWS * D_MODEL * 2);
  __bf16* og  = (__bf16*)take((size_t)ROWS * D_MODEL * 2);
  __bf16* qb  = (__bf16*)take((size_t)ROWS * D_MODEL * 2);
  __bf16* kb  = (__bf16*)take((size_t)ROWS * D_MODEL * 2);
  __bf16* vb  = (__bf16*)take((size_t)ROWS * D_MODEL * 2);
  float*  xg  = (float*)take((size_t)ROWS * D_MODEL * 4);
  // hmid is live only after q/k/v are dead: alias over q/k/v region (24MB >= 23.1MB)
  __bf16* hmid = qb;

  dim3 blk(256);
  long t;

  // Weight conversion (q scale dh^-0.5 = 0.125 folded into wq).
  t = (long)D_MODEL * D_MODEL;
  cvtT_kernel<<<(t + 255) / 256, blk, 0, stream>>>(wq, wqT, D_MODEL, D_MODEL, D_MODEL, 0.125f, t);
  cvtT_kernel<<<(t + 255) / 256, blk, 0, stream>>>(wk, wkT, D_MODEL, D_MODEL, D_MODEL, 1.0f, t);
  cvtT_kernel<<<(t + 255) / 256, blk, 0, stream>>>(wv, wvT, D_MODEL, D_MODEL, D_MODEL, 1.0f, t);
  cvtT_kernel<<<(t + 255) / 256, blk, 0, stream>>>(wg, wgT, D_MODEL, D_MODEL, D_MODEL, 1.0f, t);
  cvtT_kernel<<<(t + 255) / 256, blk, 0, stream>>>(wo, woT, D_MODEL, D_MODEL, D_MODEL, 1.0f, t);
  t = (long)HID2 * D_MODEL;
  cvtT_kernel<<<(t + 255) / 256, blk, 0, stream>>>(w1, w1T, D_MODEL, HID, D_MODEL, 1.0f, t);
  cvtT_kernel<<<(t + 255) / 256, blk, 0, stream>>>(w2, w2T, D_MODEL, HID, D_MODEL, 1.0f, t);
  t = (long)D_MODEL * HID2;
  cvtT_kernel<<<(t + 255) / 256, blk, 0, stream>>>(w3, w3T, HID, D_MODEL, HID2, 1.0f, t);

  // LN1 -> bf16
  ln_kernel<<<ROWS, blk, 0, stream>>>(x, ln1g, ln1b, xn);

  // Projections (bf16 WMMA, fp32 accum)
  dim3 gproj(D_MODEL / 128, ROWS / 128);   // (8, 32)
  gemm_kernel<EPI_BF16><<<gproj, blk, 0, stream>>>(xn, wqT, nullptr, qb, D_MODEL, D_MODEL);
  gemm_kernel<EPI_BF16><<<gproj, blk, 0, stream>>>(xn, wkT, nullptr, kb, D_MODEL, D_MODEL);
  gemm_kernel<EPI_BF16><<<gproj, blk, 0, stream>>>(xn, wvT, nullptr, vb, D_MODEL, D_MODEL);
  gemm_kernel<EPI_F32 ><<<gproj, blk, 0, stream>>>(xn, wgT, nullptr, xg, D_MODEL, D_MODEL);

  // Retention (decay-masked attention + GroupNorm + SiLU gate)
  dim3 gret(SEQ / 128, NHEAD, BATCH);
  retention_kernel<<<gret, blk, 0, stream>>>(qb, kb, vb, xg, og);

  // y = x + gated @ wo   (writes d_out)
  gemm_kernel<EPI_RES><<<gproj, blk, 0, stream>>>(og, woT, x, out, D_MODEL, D_MODEL);

  // LN2 -> bf16 (reuse xn)
  ln_kernel<<<ROWS, blk, 0, stream>>>(out, ln2g, ln2b, xn);

  // hmid = silu(xn@w1) * (xn@w2)  (dual-B WMMA GEMM, padded N)
  dim3 gsw(HID2 / 64, ROWS / 128);         // (44, 32)
  swiglu_kernel<<<gsw, blk, 0, stream>>>(xn, w1T, w2T, hmid, HID2, D_MODEL);

  // out = y + hmid @ w3  (residual read/write on d_out, per-element ownership)
  gemm_kernel<EPI_RES><<<gproj, blk, 0, stream>>>(hmid, w3T, out, out, D_MODEL, HID2);

  (void)in_sizes; (void)n_in; (void)out_size; (void)ws_size;
}
